// SA_WFFCM_24378234372600
// MI455X (gfx1250) — compile-verified, matrix operand order checked by reading, not run on hardware
//
#include <hip/hip_runtime.h>
#include <hip/hip_bf16.h>
#include <math.h>

namespace {

constexpr int   CH    = 192;      // IN_CH == MID
constexpr int   HW    = 65536;    // 256*256
constexpr int   WW    = 256;
constexpr int   NB    = 4;
constexpr int   RED   = 48;
constexpr int   G     = 6;
constexpr int   GC    = 32;
constexpr int   LDK   = 40;       // padded f16 stride of a 32-wide K slice (bank-friendly)
constexpr int   NTILE = 128;      // pixels per block
constexpr float EPSB  = 1e-5f;

typedef __attribute__((ext_vector_type(16))) _Float16 v16h;
typedef __attribute__((ext_vector_type(8)))  _Float16 v8h;
typedef __attribute__((ext_vector_type(8)))  float    v8f;

__device__ __forceinline__ float gelu_exact(float v) {
  return 0.5f * v * (1.0f + erff(v * 0.70710678118654752f));
}
__device__ __forceinline__ v16h cat8(v8h lo, v8h hi) {
  return __builtin_shufflevector(lo, hi, 0,1,2,3,4,5,6,7,8,9,10,11,12,13,14,15);
}

enum { EPI_BIAS = 0, EPI_BN = 1, EPI_BN_GELU = 2, EPI_RESID = 3 };

// D[o,n] = sum_k W[o,k] * X[k,n] (+ optional second addend / per-k scale at staging),
// fused epilogue. M=K=192, one block handles NTILE pixels of one batch image.
// TX: activation element type in HBM (float for the first GEMM, _Float16 after),
// TY: output element type (_Float16 for intermediates, float for the final output).
template <int EPI, typename TX, typename TY>
__global__ __launch_bounds__(256)
void gemm192_wmma(const TX* __restrict__ X1, const _Float16* __restrict__ X2,
                  const float* __restrict__ xscale,
                  const float* __restrict__ W,   const float* __restrict__ bias,
                  const float* __restrict__ bng, const float* __restrict__ bnb,
                  const float* __restrict__ bnm, const float* __restrict__ bnv,
                  const float* __restrict__ resid,
                  TY* __restrict__ Y) {
  __shared__ alignas(16) _Float16 sW[CH * LDK];     // weight K-slice, row-major [o][k]
  __shared__ alignas(16) _Float16 sX[NTILE * LDK];  // activation slice transposed [n][k]

  const int tid  = threadIdx.x;
  const int b    = blockIdx.y;
  const int nb   = blockIdx.x * NTILE;
  const int wv   = tid >> 5;             // 8 waves; wave wv owns n-tile wv
  const int lane = tid & 31;
  const int row  = lane & 15;
  const int kb   = (lane >> 4) << 3;     // 0 or 8 : K-base of fragment, also M-offset of D

  const TX*       Xb1 = X1 + (size_t)b * CH * HW;
  const _Float16* Xb2 = X2 ? (X2 + (size_t)b * CH * HW) : nullptr;

  const v8f vzero = {0.f, 0.f, 0.f, 0.f, 0.f, 0.f, 0.f, 0.f};
  v8f acc[12];
#pragma unroll
  for (int j = 0; j < 12; ++j) acc[j] = vzero;

  for (int kk = 0; kk < CH; kk += 32) {
    __syncthreads();
    // Stage W slice (coalesced along k; W totals 147 KB so repeats are L2-resident)
    for (int i = tid; i < CH * 32; i += 256) {
      int o = i >> 5, k = i & 31;
      sW[o * LDK + k] = (_Float16)W[o * CH + kk + k];
    }
    // Stage X slice transposed (coalesced along n)
    for (int i = tid; i < NTILE * 32; i += 256) {
      int n = i & (NTILE - 1), k = i >> 7;
      float v = (float)Xb1[(size_t)(kk + k) * HW + nb + n];
      if (Xb2) v += (float)Xb2[(size_t)(kk + k) * HW + nb + n];
      if constexpr (EPI == EPI_RESID) v *= xscale[b * CH + kk + k];  // SE channel scale
      sX[n * LDK + k] = (_Float16)v;
    }
    __syncthreads();

    // B fragment: column n = wv*16+row, K per ISA 16-bit layout (two ds_read_b128)
    const _Float16* bp = &sX[(wv * 16 + row) * LDK + kb];
    v16h bf = cat8(*(const v8h*)bp, *(const v8h*)(bp + 16));

    // Two batches of 6 A fragments in distinct registers so the DS pipe can run
    // ahead of the matrix pipe (one s_wait_dscnt per 6 WMMAs, not per WMMA).
    v16h af[6];
#pragma unroll
    for (int j = 0; j < 6; ++j) {
      const _Float16* ap = &sW[(j * 16 + row) * LDK + kb];
      af[j] = cat8(*(const v8h*)ap, *(const v8h*)(ap + 16));
    }
#pragma unroll
    for (int j = 0; j < 6; ++j)
      acc[j] = __builtin_amdgcn_wmma_f32_16x16x32_f16(false, af[j], false, bf,
                                                      (short)0, acc[j], false, false);
#pragma unroll
    for (int j = 0; j < 6; ++j) {
      const _Float16* ap = &sW[((j + 6) * 16 + row) * LDK + kb];
      af[j] = cat8(*(const v8h*)ap, *(const v8h*)(ap + 16));
    }
#pragma unroll
    for (int j = 0; j < 6; ++j)
      acc[j + 6] = __builtin_amdgcn_wmma_f32_16x16x32_f16(false, af[j], false, bf,
                                                          (short)0, acc[j + 6], false, false);
  }

  const int ncol = nb + wv * 16 + row;
#pragma unroll
  for (int j = 0; j < 12; ++j) {
#pragma unroll
    for (int r = 0; r < 8; ++r) {
      int o = j * 16 + r + kb;           // D layout: m = r + 8*(lane>=16)
      float v = acc[j][r] + bias[o];
      if constexpr (EPI == EPI_BN || EPI == EPI_BN_GELU) {
        float sc = bng[o] * rsqrtf(bnv[o] + EPSB);
        v = (v - bnm[o]) * sc + bnb[o];
      }
      if constexpr (EPI == EPI_BN_GELU) v = gelu_exact(v);
      if constexpr (EPI == EPI_RESID)
        v += resid[(size_t)(b * CH + o) * HW + ncol];
      Y[(size_t)(b * CH + o) * HW + ncol] = (TY)v;
    }
  }
}

// Depthwise 3x3 (zero pad) + bias + BN1 + GELU, f16 in / f16 out (f32 math)
__global__ __launch_bounds__(256)
void dw_bn_gelu(const _Float16* __restrict__ mid, const float* __restrict__ dww,
                const float* __restrict__ dwb,
                const float* __restrict__ g1, const float* __restrict__ b1,
                const float* __restrict__ m1, const float* __restrict__ v1,
                _Float16* __restrict__ out) {
  size_t idx = (size_t)blockIdx.x * 256 + threadIdx.x;
  int x = (int)(idx & 255);
  int y = (int)((idx >> 8) & 255);
  int rest = (int)(idx >> 16);
  int c = rest % CH;
  const _Float16* p  = mid + ((size_t)rest << 16);
  const float*    wp = dww + c * 9;
  float acc = 0.f;
#pragma unroll
  for (int dy = -1; dy <= 1; ++dy) {
    int yy = y + dy;
    if (yy < 0 || yy > 255) continue;
#pragma unroll
    for (int dx = -1; dx <= 1; ++dx) {
      int xx = x + dx;
      if (xx < 0 || xx > 255) continue;
      acc += wp[(dy + 1) * 3 + (dx + 1)] * (float)p[(yy << 8) + xx];
    }
  }
  float sc = g1[c] * rsqrtf(v1[c] + EPSB);
  float v  = (acc + dwb[c] - m1[c]) * sc + b1[c];
  out[idx] = (_Float16)gelu_exact(v);
}

__constant__ float C8c[8] = {1.f, 0.70710678118654752f, 0.f, -0.70710678118654752f,
                             -1.f, -0.70710678118654752f, 0.f, 0.70710678118654752f};
__constant__ float S8c[8] = {0.f, 0.70710678118654752f, 1.f, 0.70710678118654752f,
                             0.f, -0.70710678118654752f, -1.f, -0.70710678118654752f};

// Windowed rfft2 -> bin-independent 64x64 channel mix -> irfft2 -> fBN.
// One 64-thread block per (batch, group, 8x8 window). 256%8==0 so no padding.
__global__ __launch_bounds__(64)
void fft_mix(const _Float16* __restrict__ mid, const float* __restrict__ fw,
             const float* __restrict__ fb,
             const float* __restrict__ gp, const float* __restrict__ bp,
             const float* __restrict__ mp, const float* __restrict__ vp,
             _Float16* __restrict__ fout) {
  __shared__ float tile[GC][64];
  __shared__ float Ar[GC][40], Ai[GC][40];
  __shared__ float Br[GC][40], Bi[GC][40];
  __shared__ float Wg[64 * 64];
  __shared__ float fbias[64];

  const int tid = threadIdx.x;
  int bid = blockIdx.x;
  int wx = bid & 31, wy = (bid >> 5) & 31;
  int rest = bid >> 10;
  int g = rest % G, b = rest / G;

  for (int i = tid; i < 64 * 64; i += 64) Wg[i] = fw[g * 4096 + i];
  fbias[tid] = fb[g * 64 + tid];
  for (int i = tid; i < GC * 64; i += 64) {
    int c = i >> 6, p = i & 63;
    int y = p >> 3, x = p & 7;
    tile[c][p] = (float)mid[(size_t)(b * CH + g * GC + c) * HW + (wy * 8 + y) * WW + wx * 8 + x];
  }
  __syncthreads();
  // Row real DFT (x -> kx=0..4): Ar/Ai[c][y*5+kx]
  for (int t = tid; t < GC * 40; t += 64) {
    int kx = t % 5, y = (t / 5) & 7, c = t / 40;
    float re = 0.f, im = 0.f;
    for (int x = 0; x < 8; ++x) {
      float v = tile[c][y * 8 + x];
      int ph = (kx * x) & 7;
      re += v * C8c[ph];
      im -= v * S8c[ph];
    }
    Ar[c][y * 5 + kx] = re; Ai[c][y * 5 + kx] = im;
  }
  __syncthreads();
  // Column complex DFT (y -> ky), ortho 1/8: Br/Bi[c][ky*5+kx]
  for (int t = tid; t < GC * 40; t += 64) {
    int kx = t % 5, ky = (t / 5) & 7, c = t / 40;
    float re = 0.f, im = 0.f;
    for (int y = 0; y < 8; ++y) {
      float a = Ar[c][y * 5 + kx], bb2 = Ai[c][y * 5 + kx];
      int ph = (ky * y) & 7;
      re += a * C8c[ph] + bb2 * S8c[ph];
      im += bb2 * C8c[ph] - a * S8c[ph];
    }
    Br[c][ky * 5 + kx] = 0.125f * re; Bi[c][ky * 5 + kx] = 0.125f * im;
  }
  __syncthreads();
  // Channel mixing (same 64x64 matrix for every bin) + bias -> Ar/Ai
  for (int t = tid; t < GC * 40; t += 64) {
    int bin = t % 40, o = t / 40;
    float gr = fbias[o], gi = fbias[o + GC];
    const float* w0 = &Wg[o * 64];
    const float* w1 = &Wg[(o + GC) * 64];
    for (int c = 0; c < GC; ++c) {
      float fr = Br[c][bin], fi = Bi[c][bin];
      gr += w0[c] * fr + w0[GC + c] * fi;
      gi += w1[c] * fr + w1[GC + c] * fi;
    }
    Ar[o][bin] = gr; Ai[o][bin] = gi;
  }
  __syncthreads();
  // Inverse column DFT (ky -> y): Br/Bi[o][y*5+kx]
  for (int t = tid; t < GC * 40; t += 64) {
    int kx = t % 5, y = (t / 5) & 7, o = t / 40;
    float re = 0.f, im = 0.f;
    for (int ky = 0; ky < 8; ++ky) {
      float a = Ar[o][ky * 5 + kx], bb2 = Ai[o][ky * 5 + kx];
      int ph = (ky * y) & 7;
      re += a * C8c[ph] - bb2 * S8c[ph];
      im += bb2 * C8c[ph] + a * S8c[ph];
    }
    Br[o][y * 5 + kx] = re; Bi[o][y * 5 + kx] = im;
  }
  __syncthreads();
  // Inverse row (kx -> x, Hermitian half, Im of bins 0/4 ignored) + ortho 1/8 + fBN + store
  for (int t = tid; t < GC * 8; t += 64) {
    int o = t >> 3, y = t & 7;
    float h0 = Br[o][y * 5 + 0], h4 = Br[o][y * 5 + 4];
    float hr1 = Br[o][y * 5 + 1], hi1 = Bi[o][y * 5 + 1];
    float hr2 = Br[o][y * 5 + 2], hi2 = Bi[o][y * 5 + 2];
    float hr3 = Br[o][y * 5 + 3], hi3 = Bi[o][y * 5 + 3];
    int ch = g * GC + o;
    float sc = gp[ch] * rsqrtf(vp[ch] + EPSB);
    float sh = bp[ch] - mp[ch] * sc;
    size_t base = (size_t)(b * CH + ch) * HW + (wy * 8 + y) * WW + wx * 8;
    for (int x = 0; x < 8; ++x) {
      float v = h0 + ((x & 1) ? -h4 : h4);
      int p1 = x & 7, p2 = (2 * x) & 7, p3 = (3 * x) & 7;
      v += 2.f * (hr1 * C8c[p1] - hi1 * S8c[p1]);
      v += 2.f * (hr2 * C8c[p2] - hi2 * S8c[p2]);
      v += 2.f * (hr3 * C8c[p3] - hi3 * S8c[p3]);
      v *= 0.125f;
      fout[base + x] = (_Float16)(v * sc + sh);
    }
  }
}

// Global average pool over f16 rows: one block per (b, c) row, v8h vector loads
__global__ __launch_bounds__(256)
void gap_kernel(const _Float16* __restrict__ fused, float* __restrict__ gap) {
  __shared__ float red[256];
  int rowid = blockIdx.x;
  const v8h* p = (const v8h*)(fused + (size_t)rowid * HW);
  float s = 0.f;
  for (int i = threadIdx.x; i < HW / 8; i += 256) {
    v8h v = p[i];
    float t = 0.f;
#pragma unroll
    for (int j = 0; j < 8; ++j) t += (float)v[j];
    s += t;
  }
  red[threadIdx.x] = s;
  __syncthreads();
  for (int st = 128; st > 0; st >>= 1) {
    if (threadIdx.x < st) red[threadIdx.x] += red[threadIdx.x + st];
    __syncthreads();
  }
  if (threadIdx.x == 0) gap[rowid] = red[0] * (1.f / (float)HW);
}

// SE MLP: gap[4,192] -> sigmoid scales se[4,192]
__global__ __launch_bounds__(CH)
void se_kernel(const float* __restrict__ gap, const float* __restrict__ w1,
               const float* __restrict__ b1, const float* __restrict__ w2,
               const float* __restrict__ b2, float* __restrict__ se) {
  __shared__ float hid[RED];
  int t = threadIdx.x;
  for (int b = 0; b < NB; ++b) {
    if (t < RED) {
      float a = b1[t];
      for (int i = 0; i < CH; ++i) a += gap[b * CH + i] * w1[t * CH + i];
      hid[t] = gelu_exact(a);
    }
    __syncthreads();
    float a = b2[t];
    for (int r = 0; r < RED; ++r) a += hid[r] * w2[t * RED + r];
    se[b * CH + t] = 1.f / (1.f + expf(-a));
    __syncthreads();
  }
}

}  // namespace

extern "C" void kernel_launch(void* const* d_in, const int* in_sizes, int n_in,
                              void* d_out, int out_size, void* d_ws, size_t ws_size,
                              hipStream_t stream) {
  const float* x        = (const float*)d_in[0];
  const float* reduce_w = (const float*)d_in[1];
  const float* reduce_b = (const float*)d_in[2];
  const float* dw_w     = (const float*)d_in[3];
  const float* dw_b     = (const float*)d_in[4];
  const float* bn1_g    = (const float*)d_in[5];
  const float* bn1_b    = (const float*)d_in[6];
  const float* bn1_m    = (const float*)d_in[7];
  const float* bn1_v    = (const float*)d_in[8];
  const float* pw_w     = (const float*)d_in[9];
  const float* pw_b     = (const float*)d_in[10];
  const float* bn2_g    = (const float*)d_in[11];
  const float* bn2_b    = (const float*)d_in[12];
  const float* bn2_m    = (const float*)d_in[13];
  const float* bn2_v    = (const float*)d_in[14];
  const float* freq_w   = (const float*)d_in[15];
  const float* freq_b   = (const float*)d_in[16];
  const float* fbn_g    = (const float*)d_in[17];
  const float* fbn_b    = (const float*)d_in[18];
  const float* fbn_m    = (const float*)d_in[19];
  const float* fbn_v    = (const float*)d_in[20];
  const float* cross_w  = (const float*)d_in[21];
  const float* cross_b  = (const float*)d_in[22];
  const float* cbn_g    = (const float*)d_in[23];
  const float* cbn_b    = (const float*)d_in[24];
  const float* cbn_m    = (const float*)d_in[25];
  const float* cbn_v    = (const float*)d_in[26];
  const float* fc1_w    = (const float*)d_in[27];
  const float* fc1_b    = (const float*)d_in[28];
  const float* fc2_w    = (const float*)d_in[29];
  const float* fc2_b    = (const float*)d_in[30];
  const float* out_w    = (const float*)d_in[31];
  const float* out_b    = (const float*)d_in[32];

  const size_t NE = (size_t)NB * CH * HW;
  _Float16* buf0 = (_Float16*)d_ws;     // mid, later fused        (f16)
  _Float16* buf1 = buf0 + NE;           // dw/gelu out, later fft  (f16)
  _Float16* buf2 = buf1 + NE;           // pw/bn2 out (s)          (f16)
  float*    gapb = (float*)(buf2 + NE); // [4*192]
  float*    seb  = gapb + NB * CH;      // [4*192]

  dim3 gg(HW / NTILE, NB);              // 512 x 4 blocks, 256 threads

  // 1. mid = reduce_w @ x + reduce_b        (f32 in, f16 out)
  gemm192_wmma<EPI_BIAS, float, _Float16><<<gg, 256, 0, stream>>>(
      x, nullptr, nullptr, reduce_w, reduce_b,
      nullptr, nullptr, nullptr, nullptr, nullptr, buf0);
  // 2. t = gelu(bn1(dw3x3(mid)))
  dw_bn_gelu<<<(int)(NE / 256), 256, 0, stream>>>(
      buf0, dw_w, dw_b, bn1_g, bn1_b, bn1_m, bn1_v, buf1);
  // 3. s = bn2(pw_w @ t + pw_b)
  gemm192_wmma<EPI_BN, _Float16, _Float16><<<gg, 256, 0, stream>>>(
      buf1, nullptr, nullptr, pw_w, pw_b,
      bn2_g, bn2_b, bn2_m, bn2_v, nullptr, buf2);
  // 4. f = fbn(windowed spectral mix of mid)
  fft_mix<<<NB * G * 32 * 32, 64, 0, stream>>>(
      buf0, freq_w, freq_b, fbn_g, fbn_b, fbn_m, fbn_v, buf1);
  // 5. fused = gelu(cbn(cross_w @ (s + f) + cross_b))
  gemm192_wmma<EPI_BN_GELU, _Float16, _Float16><<<gg, 256, 0, stream>>>(
      buf2, buf1, nullptr, cross_w, cross_b,
      cbn_g, cbn_b, cbn_m, cbn_v, nullptr, buf0);
  // 6-7. SE gates
  gap_kernel<<<NB * CH, 256, 0, stream>>>(buf0, gapb);
  se_kernel<<<1, CH, 0, stream>>>(gapb, fc1_w, fc1_b, fc2_w, fc2_b, seb);
  // 8. out = x + out_w @ (fused * se) + out_b   (SE scale folded into X staging; f32 out)
  gemm192_wmma<EPI_RESID, _Float16, float><<<gg, 256, 0, stream>>>(
      buf0, nullptr, seb, out_w, out_b,
      nullptr, nullptr, nullptr, nullptr, x, (float*)d_out);
}
